// DebugLinear_82051055223325
// MI455X (gfx1250) — matrix operand order repaired and checked by reading, not compiled, not run on hardware
//
#include <hip/hip_runtime.h>
#include <stdint.h>

// ---------------- problem constants ----------------
#define N_TOK 8192   // rows of x  (GEMM M)
#define D_K   4096   // in_features (GEMM K)
#define D_N   4096   // out_features (GEMM N)

#define BM 128
#define BN 128
#define BK 32
#define LDBF 40      // bf16 elements per LDS row (32 + 8 pad): 20 dwords, 5 coprime 16
                     // -> b128 frag reads / stage writes are bank-conflict free

typedef __attribute__((ext_vector_type(16))) __bf16 v16bf;
typedef __attribute__((ext_vector_type(8)))  float  v8f;
typedef __attribute__((ext_vector_type(4)))  float  v4f;

union FragU { v16bf bf; v4f q[2]; };          // 32B: one WMMA A/B fragment per lane
union PackU { __bf16 h[16]; v4f q[2]; };      // 16 bf16 packed for two b128 stores

// Split fp32 -> bf16 (hi, lo), RNE; hi + lo captures f to ~2^-16 relative.
__device__ __forceinline__ void split_bf16(float f, __bf16& h, __bf16& l) {
  h = (__bf16)f;
  l = (__bf16)(f - (float)h);
}

struct Stage { v4f a[4]; v4f b[4]; };   // one thread's half-row of A and of B (16 floats each)

__device__ __forceinline__ Stage gload(const float* __restrict__ ga,
                                       const float* __restrict__ gb,
                                       int row, int kofs) {
  Stage s;
  const float* pa = ga + (size_t)row * D_K + kofs;
  const float* pb = gb + (size_t)row * D_K + kofs;
#pragma unroll
  for (int i = 0; i < 4; ++i) {
    s.a[i] = *(const v4f*)(pa + 4 * i);
    s.b[i] = *(const v4f*)(pb + 4 * i);
  }
  return s;
}

// Convert one thread's staged registers and store bf16 hi/lo tiles to LDS.
__device__ __forceinline__ void cvt_store(const Stage& s, float th,
                                          __bf16* __restrict__ ahi, __bf16* __restrict__ alo,
                                          __bf16* __restrict__ bhi, __bf16* __restrict__ blo,
                                          int row, int kofs) {
  PackU ah, al, bh, bl;
#pragma unroll
  for (int i = 0; i < 4; ++i) {
#pragma unroll
    for (int e = 0; e < 4; ++e) {
      float va = s.a[i][e];
      va = (__builtin_fabsf(va) >= th) ? va : 0.0f;   // runtime threshold pruning (A only)
      split_bf16(va, ah.h[i * 4 + e], al.h[i * 4 + e]);
      split_bf16(s.b[i][e], bh.h[i * 4 + e], bl.h[i * 4 + e]);
    }
  }
  const int o = row * LDBF + kofs;
  *(v4f*)(ahi + o) = ah.q[0];  *(v4f*)(ahi + o + 8) = ah.q[1];
  *(v4f*)(alo + o) = al.q[0];  *(v4f*)(alo + o + 8) = al.q[1];
  *(v4f*)(bhi + o) = bh.q[0];  *(v4f*)(bhi + o + 8) = bh.q[1];
  *(v4f*)(blo + o) = bl.q[0];  *(v4f*)(blo + o + 8) = bl.q[1];
}

// ---------------- GEMM with runtime threshold pruning (bf16x3) ----------------
__global__ __launch_bounds__(256)
void gemm_prune_kernel(const float* __restrict__ x, const float* __restrict__ w,
                       const float* __restrict__ thp, float* __restrict__ out) {
  // bf16 hi/lo tiles, double-buffered: 8 * 128*40*2B = 80 KB
  __shared__ __align__(16) __bf16 Ahi[2][BM * LDBF];
  __shared__ __align__(16) __bf16 Alo[2][BM * LDBF];
  __shared__ __align__(16) __bf16 Bhi[2][BN * LDBF];
  __shared__ __align__(16) __bf16 Blo[2][BN * LDBF];

  const int t      = threadIdx.x;
  const int lane   = t & 31;
  const int wv     = t >> 5;             // wave 0..7
  const int wm     = wv & 3;             // 4 waves along M -> 32 rows each
  const int wn     = wv >> 2;            // 2 waves along N -> 64 cols each
  const int l15    = lane & 15;
  const int half8  = (lane >> 4) << 3;   // A frag K-base: 0 / 8
  const int half16 = (lane >> 4) << 4;   // unused; B frags also use half8 (0 / 8)

  const int srow  = t >> 1;              // staging: row handled by this thread
  const int skofs = (t & 1) << 4;        // staging: K offset (0 or 16)

  const int n0 = blockIdx.x * BN;
  const int m0 = blockIdx.y * BM;

  const float th = thp[0];

  const float* ga = x + (size_t)m0 * D_K;   // A panel [128 x K]
  const float* gb = w + (size_t)n0 * D_K;   // W panel [128 x K] (row-major)

  v8f acc[2][4] = {};

  // Prologue: stage tile 0, start loads for tile 1.
  Stage regs = gload(ga, gb, srow, skofs);
  cvt_store(regs, th, Ahi[0], Alo[0], Bhi[0], Blo[0], srow, skofs);
  regs = gload(ga, gb, srow, BK + skofs);
  __syncthreads();

  const int ksteps = D_K / BK;   // 128
  for (int kt = 0; kt < ksteps; ++kt) {
    const int cur = kt & 1;

    if (kt + 1 < ksteps) {
      const int nxt = cur ^ 1;
      cvt_store(regs, th, Ahi[nxt], Alo[nxt], Bhi[nxt], Blo[nxt], srow, skofs);
      if (kt + 2 < ksteps)
        regs = gload(ga, gb, srow, (kt + 2) * BK + skofs);
      if (kt + 4 < ksteps) {
        __builtin_prefetch(ga + (size_t)srow * D_K + (kt + 4) * BK + skofs, 0, 1);
        __builtin_prefetch(gb + (size_t)srow * D_K + (kt + 4) * BK + skofs, 0, 1);
      }
    }

    // ---- A fragments: 16x32 bf16 layout (2 K per VGPR).
    // lane<16 : K 0..7 then 16..23 ; lane>=16: K 8..15 then 24..31
    FragU ah[2], al[2];
#pragma unroll
    for (int i = 0; i < 2; ++i) {
      const int r = (wm * 32 + i * 16 + l15) * LDBF;
      ah[i].q[0] = *(const v4f*)(&Ahi[cur][r + half8]);
      ah[i].q[1] = *(const v4f*)(&Ahi[cur][r + 16 + half8]);
      al[i].q[0] = *(const v4f*)(&Alo[cur][r + half8]);
      al[i].q[1] = *(const v4f*)(&Alo[cur][r + 16 + half8]);
    }

    // ---- B fragments + 3 WMMAs per (i,j): D += Ah*Bh + Ah*Bl + Al*Bh
    // B 32x16 bf16 layout: lane = column; lane<16 holds K0..7,16..23, lane>=16 K8..15,24..31.
#pragma unroll
    for (int j = 0; j < 4; ++j) {
      const int r = (wn * 64 + j * 16 + l15) * LDBF;
      FragU bh, bl;
      bh.q[0] = *(const v4f*)(&Bhi[cur][r + half8]);
      bh.q[1] = *(const v4f*)(&Bhi[cur][r + 16 + half8]);
      bl.q[0] = *(const v4f*)(&Blo[cur][r + half8]);
      bl.q[1] = *(const v4f*)(&Blo[cur][r + 16 + half8]);
#pragma unroll
      for (int i = 0; i < 2; ++i) {
        acc[i][j] = __builtin_amdgcn_wmma_f32_16x16x32_bf16(
            false, ah[i].bf, false, bh.bf, (short)0, acc[i][j], false, false);
        acc[i][j] = __builtin_amdgcn_wmma_f32_16x16x32_bf16(
            false, ah[i].bf, false, bl.bf, (short)0, acc[i][j], false, false);
        acc[i][j] = __builtin_amdgcn_wmma_f32_16x16x32_bf16(
            false, al[i].bf, false, bh.bf, (short)0, acc[i][j], false, false);
      }
    }

    __syncthreads();
  }

  // Epilogue: C/D layout — VGPR r: lanes 0-15 = row M=r, lanes 16-31 = row M=8+r.
#pragma unroll
  for (int i = 0; i < 2; ++i) {
#pragma unroll
    for (int j = 0; j < 4; ++j) {
      const int colg = n0 + wn * 64 + j * 16 + l15;
      const int rowb = m0 + wm * 32 + i * 16 + ((lane >> 4) << 3);
#pragma unroll
      for (int r = 0; r < 8; ++r)
        out[(size_t)(rowb + r) * D_N + colg] = acc[i][j][r];
    }
  }
}

// ---------------- pruned-count pass (memory bound) ----------------
__global__ void zero_counter_kernel(unsigned int* c) { *c = 0u; }

__global__ __launch_bounds__(256)
void prune_count_kernel(const float* __restrict__ x, const float* __restrict__ thp,
                        unsigned int* __restrict__ cnt) {
  __shared__ unsigned int lcnt;
  if (threadIdx.x == 0) lcnt = 0u;
  __syncthreads();

  const float th = thp[0];
  unsigned int c = 0;
  const size_t n4 = (size_t)N_TOK * D_K / 4;
  const size_t stride = (size_t)gridDim.x * blockDim.x;
  const float4* x4 = (const float4*)x;
  for (size_t i = (size_t)blockIdx.x * blockDim.x + threadIdx.x; i < n4; i += stride) {
    float4 v = x4[i];
    c += (__builtin_fabsf(v.x) >= th) ? 1u : 0u;
    c += (__builtin_fabsf(v.y) >= th) ? 1u : 0u;
    c += (__builtin_fabsf(v.z) >= th) ? 1u : 0u;
    c += (__builtin_fabsf(v.w) >= th) ? 1u : 0u;
  }
  atomicAdd(&lcnt, c);
  __syncthreads();
  if (threadIdx.x == 0) atomicAdd(cnt, lcnt);
}

__global__ void finalize_counts_kernel(const unsigned int* __restrict__ cnt,
                                       float* __restrict__ tail) {
  const unsigned long long total = (unsigned long long)N_TOK * D_K;
  unsigned long long kept = *cnt;
  // Tuple outputs concatenate after C; int32 scalars promote to f32.
  tail[0] = (float)(total - kept);  // pruned
  tail[1] = (float)total;           // total
}

// ---------------- launch ----------------
extern "C" void kernel_launch(void* const* d_in, const int* in_sizes, int n_in,
                              void* d_out, int out_size, void* d_ws, size_t ws_size,
                              hipStream_t stream) {
  const float* x  = (const float*)d_in[0];   // [8192, 4096]
  const float* w  = (const float*)d_in[1];   // [4096, 4096]
  const float* th = (const float*)d_in[2];   // [1]
  float* out = (float*)d_out;
  unsigned int* cnt = (unsigned int*)d_ws;

  zero_counter_kernel<<<1, 1, 0, stream>>>(cnt);
  prune_count_kernel<<<4096, 256, 0, stream>>>(x, th, cnt);
  finalize_counts_kernel<<<1, 1, 0, stream>>>(cnt, out + (size_t)N_TOK * D_N);

  dim3 grid(D_N / BN, N_TOK / BM);   // 32 x 64 workgroups
  gemm_prune_kernel<<<grid, 256, 0, stream>>>(x, w, th, out);
}
